// MLAAttention_8272107012693
// MI455X (gfx1250) — matrix operand order repaired and checked by reading, not compiled, not run on hardware
//
#include <hip/hip_runtime.h>

// ---------------------------------------------------------------------------
// MLA attention forward for MI455X (gfx1250), bf16 WMMA compute / fp32 accum.
// TDM (tensor_load_to_lds) staging for attention K/V tiles.
// ---------------------------------------------------------------------------

typedef __bf16 bf16;
typedef __attribute__((ext_vector_type(16))) __bf16 v16bf;
typedef __attribute__((ext_vector_type(8)))  float  v8f;
typedef __attribute__((ext_vector_type(4)))  unsigned int u32x4;
typedef __attribute__((ext_vector_type(8)))  int          i32x8;
typedef __attribute__((ext_vector_type(4)))  int          i32x4;

#define B_    2
#define S_    2048
#define H_    2048
#define NH_   16
#define HD_   128
#define RD_   64

#if __has_builtin(__builtin_amdgcn_tensor_load_to_lds) && \
    __has_builtin(__builtin_amdgcn_s_wait_tensorcnt)
#define USE_TDM 1
#else
#define USE_TDM 0
#endif

__device__ __forceinline__ bf16 to_bf16(float f) {
    union { float f; unsigned u; } x; x.f = f;
    unsigned r = x.u + 0x7fffu + ((x.u >> 16) & 1u);   // round-to-nearest-even
    unsigned short h = (unsigned short)(r >> 16);
    return __builtin_bit_cast(bf16, h);
}
__device__ __forceinline__ float bf2f(bf16 h) {
    unsigned short s = __builtin_bit_cast(unsigned short, h);
    union { unsigned u; float f; } x; x.u = ((unsigned)s) << 16;
    return x.f;
}

// Generic LDS pointer -> workgroup LDS byte address (aperture maps addr[31:0]).
__device__ __forceinline__ unsigned lds_off(const void* p) {
    return (unsigned)(unsigned long long)(p);
}

// A-fragment (16x32 bf16, M x K), ISA layout:
//  lanes 0-15 : row M=lane,   slots 0..7 <- K k0+0..7,   slots 8..15 <- K k0+16..23
//  lanes16-31 : row M=lane-16, slots 0..7 <- K k0+8..15, slots 8..15 <- K k0+24..31
__device__ __forceinline__ v16bf load_frag_a(const bf16* base, int ld) {
    int lane = threadIdx.x & 31;
    const bf16* p = base + (size_t)(lane & 15) * ld + ((lane >> 4) << 3);
    union { v16bf v; uint4 u[2]; } r;
    r.u[0] = *(const uint4*)p;
    r.u[1] = *(const uint4*)(p + 16);
    return r.v;
}

// B-fragment (32x16 bf16, K x N), sourced from Bt rows (Bt is B^T, [N][K]):
//  lanes 0-15 : col N=lane,    slots 0..7 <- K k0+0..7,  slots 8..15 <- K k0+16..23
//  lanes16-31 : col N=lane-16, slots 0..7 <- K k0+8..15, slots 8..15 <- K k0+24..31
__device__ __forceinline__ v16bf load_frag_b(const bf16* base, int ld) {
    int lane = threadIdx.x & 31;
    const bf16* p = base + (size_t)(lane & 15) * ld + ((lane >> 4) << 3);
    union { v16bf v; uint4 u[2]; } r;
    r.u[0] = *(const uint4*)p;
    r.u[1] = *(const uint4*)(p + 16);
    return r.v;
}

__device__ __forceinline__ v8f wmma_bf16(v16bf a, v16bf b, v8f c) {
    return __builtin_amdgcn_wmma_f32_16x16x32_bf16(
        false, a, false, b, (short)0, c, false, false);
}

#if USE_TDM
// Issue a 2D TDM tile load: tileY rows of tileX bf16 elements, row stride
// strideElems, into LDS at ldsByteAddr (rows stored contiguously).
// D# layout per CDNA5 ISA 8.3/8.4 (group0 128b, group1 256b; groups 2/3 zero).
// clang-23 builtin: (u32x4 g0, i32x8 g1, i32x4 g2, i32x4 g3, i32x8 ext, i32 cpol)
__device__ __forceinline__ void tdm_load_2d(unsigned ldsByteAddr, const void* gaddr,
                                            unsigned tileX, unsigned tileY,
                                            unsigned strideElems) {
    unsigned long long ga = (unsigned long long)gaddr;
    u32x4 g0;
    g0[0] = 0x1u;                                        // count=1, user mode
    g0[1] = ldsByteAddr;                                 // lds_addr (bytes)
    g0[2] = (unsigned)(ga & 0xffffffffu);                // global_addr[31:0]
    g0[3] = (unsigned)((ga >> 32) & 0x01ffffffu)         // global_addr[56:32]
          | (2u << 30);                                  // type = 2 (image)
    i32x8 g1;
    g1[0] = 0x00010000;                                  // data_size=1 (2 bytes)
    g1[1] = (int)((tileX & 0xffffu) << 16);              // tensor_dim0[15:0]
    g1[2] = (int)((tileX >> 16) | ((tileY & 0xffffu) << 16)); // dim0 hi | dim1 lo
    g1[3] = (int)((tileY >> 16) | (tileX << 16));        // dim1 hi | tile_dim0
    g1[4] = (int)(tileY & 0xffffu);                      // tile_dim1 (tile_dim2=0)
    g1[5] = (int)strideElems;                            // tensor_dim0_stride lo32
    g1[6] = 0;                                           // stride hi16 | dim1_stride lo
    g1[7] = 0;                                           // dim1_stride hi
    i32x4 z4 = {0, 0, 0, 0};
    i32x8 z8 = {0, 0, 0, 0, 0, 0, 0, 0};
    __builtin_amdgcn_tensor_load_to_lds(g0, g1, z4, z4, z8, 0);
}
#endif

// ---------------------------------------------------------------------------
// Prep kernels
// ---------------------------------------------------------------------------

__global__ __launch_bounds__(256) void f2bf_kernel(const float* __restrict__ in,
                                                   bf16* __restrict__ out, size_t n) {
    size_t i = (size_t)blockIdx.x * 256 + threadIdx.x;
    if (i < n) out[i] = to_bf16(in[i]);
}

// wt[n*K + k] = (bf16) w[k*N + n]   (transpose + convert)
__global__ __launch_bounds__(256) void wconvT_kernel(const float* __restrict__ w,
                                                     bf16* __restrict__ wt,
                                                     int K, int N) {
    size_t t = (size_t)blockIdx.x * 256 + threadIdx.x;
    if (t >= (size_t)K * N) return;
    size_t k = t % (size_t)K;
    size_t n = t / (size_t)K;
    wt[t] = to_bf16(w[k * (size_t)N + n]);
}

// vt[((b*16+h)*128 + d)*2048 + s] = v[(b*2048+s)*2048 + h*128 + d]
__global__ __launch_bounds__(256) void vtrans_kernel(const bf16* __restrict__ v,
                                                     bf16* __restrict__ vt) {
    size_t t = (size_t)blockIdx.x * 256 + threadIdx.x;   // 8,388,608 total
    int s = (int)(t & 2047);
    int d = (int)((t >> 11) & 127);
    int h = (int)((t >> 18) & 15);
    int b = (int)(t >> 22);
    vt[t] = v[((size_t)(b * S_ + s)) * (size_t)H_ + h * HD_ + d];
}

// In-place RoPE on q and k (first 64 dims of each head).
__global__ __launch_bounds__(256) void rope_kernel(bf16* __restrict__ q,
                                                   bf16* __restrict__ k) {
    size_t t = (size_t)blockIdx.x * 256 + threadIdx.x;   // B*S*NH*32
    int i = (int)(t & 31);
    int h = (int)((t >> 5) & 15);
    size_t tok = t >> 9;
    int pos = (int)(tok & (S_ - 1));
    float inv = __expf(-(float)(2 * i) * (9.210340371976184f / 64.0f));
    float ang = (float)pos * inv;
    float c = cosf(ang), s = sinf(ang);

    bf16* qp = q + tok * (size_t)H_ + h * HD_;
    float x0 = bf2f(qp[i]), x1 = bf2f(qp[i + 32]);
    qp[i]      = to_bf16(x0 * c - x1 * s);
    qp[i + 32] = to_bf16(x1 * c + x0 * s);

    bf16* kp = k + tok * (size_t)H_ + h * HD_;
    float y0 = bf2f(kp[i]), y1 = bf2f(kp[i + 32]);
    kp[i]      = to_bf16(y0 * c - y1 * s);
    kp[i + 32] = to_bf16(y1 * c + y0 * s);
}

// ---------------------------------------------------------------------------
// Tiled WMMA GEMM: C[M][N(map)] = A[M][K] * Bt[N][K]^T
//   8 waves: 4 (M) x 2 (N), wave tile 32x64 -> 128x128 per workgroup.
//   Output column remap: out_col = (n/group)*ostride + ooff + n%group
// ---------------------------------------------------------------------------

__device__ __forceinline__ void store_out(bf16*  p, float v) { *p = to_bf16(v); }
__device__ __forceinline__ void store_out(float* p, float v) { *p = v; }

template <typename OUT_T>
__global__ __launch_bounds__(256) void gemm_kernel(
    const bf16* __restrict__ A, int lda,
    const bf16* __restrict__ Bt, int ldb,
    OUT_T* __restrict__ C, int ldc,
    int M, int N, int K,
    int group, int ostride, int ooff)
{
    int lane = threadIdx.x & 31;
    int wave = threadIdx.x >> 5;
    int wm = wave & 3, wn = wave >> 2;
    int m0 = blockIdx.x * 128 + wm * 32;
    int n0 = blockIdx.y * 128 + wn * 64;

    v8f acc[2][4] = {};
    for (int k0 = 0; k0 < K; k0 += 32) {
        if (k0 + 32 < K) {  // prefetch next K tile (global_prefetch_b8)
            __builtin_prefetch(A  + (size_t)(m0 + lane) * lda + k0 + 32, 0, 1);
            __builtin_prefetch(Bt + (size_t)(n0 + lane) * ldb + k0 + 32, 0, 1);
            __builtin_prefetch(Bt + (size_t)(n0 + 32 + lane) * ldb + k0 + 32, 0, 1);
        }
        v16bf a0 = load_frag_a(A + (size_t)m0        * lda + k0, lda);
        v16bf a1 = load_frag_a(A + (size_t)(m0 + 16) * lda + k0, lda);
#pragma unroll
        for (int fn = 0; fn < 4; ++fn) {
            v16bf b = load_frag_b(Bt + (size_t)(n0 + fn * 16) * ldb + k0, ldb);
            acc[0][fn] = wmma_bf16(a0, b, acc[0][fn]);
            acc[1][fn] = wmma_bf16(a1, b, acc[1][fn]);
        }
    }

    int nlo   = lane & 15;
    int rbase = (lane >> 4) * 8;
#pragma unroll
    for (int fm = 0; fm < 2; ++fm)
#pragma unroll
        for (int fn = 0; fn < 4; ++fn)
#pragma unroll
            for (int r = 0; r < 8; ++r) {
                int m  = m0 + fm * 16 + rbase + r;
                int nl = n0 + fn * 16 + nlo;
                int col = (nl / group) * ostride + ooff + (nl % group);
                store_out(&C[(size_t)m * ldc + col], acc[fm][fn][r]);
            }
}

// ---------------------------------------------------------------------------
// Causal flash attention per (b, h): Q[2048][128] x K^T, softmax, x V.
//   grid = (S/128, B*NH). 8 waves x 16 query rows. 32-key blocks via TDM->LDS.
// ---------------------------------------------------------------------------

__global__ __launch_bounds__(256) void mla_attn_kernel(
    const bf16* __restrict__ q, const bf16* __restrict__ k,
    const bf16* __restrict__ vt, bf16* __restrict__ ao)
{
    __shared__ __align__(16) bf16 ldsK[32][HD_];     // [key][d]   8 KB
    __shared__ __align__(16) bf16 ldsVT[HD_][32];    // [d][key]   8 KB
    __shared__ __align__(16) bf16 ldsP[8][16][32];   // per-wave P 8 KB

    int lane = threadIdx.x & 31;
    int wave = threadIdx.x >> 5;
    int qb = blockIdx.x;                 // 0..15
    int bh = blockIdx.y;                 // 0..31
    int b = bh >> 4, h = bh & 15;

    int qrow0 = qb * 128 + wave * 16;
    size_t qtok = (size_t)(b * S_ + qrow0);

    v16bf qf[4];
#pragma unroll
    for (int j = 0; j < 4; ++j)
        qf[j] = load_frag_a(q + qtok * (size_t)H_ + h * HD_ + j * 32, H_);

    v8f o[8] = {};
    float mrun[8], lrun[8];
#pragma unroll
    for (int r = 0; r < 8; ++r) { mrun[r] = -3.0e38f; lrun[r] = 0.0f; }

    const float scale = 0.08838834764831845f;        // 1/sqrt(128)
    int nlo   = lane & 15;
    int rbase = (lane >> 4) * 8;

    int nkb = (qb + 1) * 4;                          // 32-key blocks (causal)
    for (int kb = 0; kb < nkb; ++kb) {
        int key0 = kb * 32;
        __syncthreads();                             // LDS free for restaging
#if USE_TDM
        if (wave == 0) {
            const bf16* ksrc = k + ((size_t)(b * S_ + key0)) * (size_t)H_ + h * HD_;
            const bf16* vsrc = vt + ((size_t)(bh * HD_)) * (size_t)S_ + key0;
            tdm_load_2d(lds_off(&ldsK[0][0]),  ksrc, HD_, 32, H_);  // 32 rows x 128
            tdm_load_2d(lds_off(&ldsVT[0][0]), vsrc, 32, HD_, S_);  // 128 rows x 32
            __builtin_amdgcn_s_wait_tensorcnt(0);
        }
#else
        {   // stage K block: 32 keys x 128 dims
            int krow  = threadIdx.x >> 3;
            int dpart = (threadIdx.x & 7) * 16;
            const bf16* src = k + ((size_t)(b * S_ + key0 + krow)) * (size_t)H_
                                + h * HD_ + dpart;
            *(uint4*)&ldsK[krow][dpart]     = *(const uint4*)src;
            *(uint4*)&ldsK[krow][dpart + 8] = *(const uint4*)(src + 8);
        }
        {   // stage V^T block: 128 dims x 32 keys
            int drow  = threadIdx.x >> 1;
            int kpart = (threadIdx.x & 1) * 16;
            const bf16* src = vt + ((size_t)(bh * HD_ + drow)) * (size_t)S_
                                 + key0 + kpart;
            *(uint4*)&ldsVT[drow][kpart]     = *(const uint4*)src;
            *(uint4*)&ldsVT[drow][kpart + 8] = *(const uint4*)(src + 8);
        }
#endif
        __syncthreads();

        // scores: S16x32 = Q(16x128) * K^T(128x32), two 16x16 C frags
        v8f s0 = {}, s1 = {};
#pragma unroll
        for (int j = 0; j < 4; ++j) {
            v16bf kb0 = load_frag_b(&ldsK[0][0]  + j * 32, HD_);
            v16bf kb1 = load_frag_b(&ldsK[16][0] + j * 32, HD_);
            s0 = wmma_bf16(qf[j], kb0, s0);
            s1 = wmma_bf16(qf[j], kb1, s1);
        }

        // causal mask + scale + row max
        float pmax[8];
#pragma unroll
        for (int r = 0; r < 8; ++r) {
            int qglob = qrow0 + rbase + r;
            float a = (key0 + nlo      <= qglob) ? s0[r] * scale : -3.0e38f;
            float c = (key0 + 16 + nlo <= qglob) ? s1[r] * scale : -3.0e38f;
            s0[r] = a; s1[r] = c;
            pmax[r] = fmaxf(a, c);
        }
        for (int off = 1; off < 16; off <<= 1)
#pragma unroll
            for (int r = 0; r < 8; ++r)
                pmax[r] = fmaxf(pmax[r], __shfl_xor(pmax[r], off, 32));

        // online softmax update
        float rsc[8];
#pragma unroll
        for (int r = 0; r < 8; ++r) {
            float mnew = fmaxf(mrun[r], pmax[r]);
            rsc[r] = __expf(mrun[r] - mnew);
            mrun[r] = mnew;
            s0[r] = __expf(s0[r] - mnew);
            s1[r] = __expf(s1[r] - mnew);
        }
        float psum[8];
#pragma unroll
        for (int r = 0; r < 8; ++r) psum[r] = s0[r] + s1[r];
        for (int off = 1; off < 16; off <<= 1)
#pragma unroll
            for (int r = 0; r < 8; ++r)
                psum[r] += __shfl_xor(psum[r], off, 32);
#pragma unroll
        for (int r = 0; r < 8; ++r) lrun[r] = lrun[r] * rsc[r] + psum[r];
#pragma unroll
        for (int j = 0; j < 8; ++j)
#pragma unroll
            for (int r = 0; r < 8; ++r) o[j][r] *= rsc[r];

        // C-layout P -> LDS -> A-layout (per-wave private, in-order LDS)
#pragma unroll
        for (int r = 0; r < 8; ++r) {
            ldsP[wave][rbase + r][nlo]      = to_bf16(s0[r]);
            ldsP[wave][rbase + r][16 + nlo] = to_bf16(s1[r]);
        }
        v16bf pa = load_frag_a(&ldsP[wave][0][0], 32);

        // O += P(16x32) * V(32x128): 8 WMMAs over d
#pragma unroll
        for (int j = 0; j < 8; ++j) {
            v16bf vb = load_frag_b(&ldsVT[j * 16][0], 32);
            o[j] = wmma_bf16(pa, vb, o[j]);
        }
    }

    // epilogue: normalize and store bf16
#pragma unroll
    for (int r = 0; r < 8; ++r) {
        float inv = 1.0f / lrun[r];
        size_t tok = qtok + rbase + r;
        bf16* dst = ao + tok * (size_t)H_ + h * HD_;
#pragma unroll
        for (int j = 0; j < 8; ++j)
            dst[j * 16 + nlo] = to_bf16(o[j][r] * inv);
    }
}

// ---------------------------------------------------------------------------
// Host orchestration
// ---------------------------------------------------------------------------

extern "C" void kernel_launch(void* const* d_in, const int* in_sizes, int n_in,
                              void* d_out, int out_size, void* d_ws, size_t ws_size,
                              hipStream_t stream) {
    const float* hs     = (const float*)d_in[0];
    const float* w_dq   = (const float*)d_in[1];
    const float* w_uq   = (const float*)d_in[2];
    const float* w_dkv  = (const float*)d_in[3];
    const float* w_uk   = (const float*)d_in[4];
    const float* w_uv   = (const float*)d_in[5];
    const float* w_kr   = (const float*)d_in[6];
    const float* w_o    = (const float*)d_in[7];
    float* out = (float*)d_out;

    const size_t T = (size_t)B_ * S_;        // 4096 tokens
    const size_t HS = T * H_;                // 8,388,608

    bf16* p = (bf16*)d_ws;
    bf16* hsb   = p; p += HS;
    bf16* wdqT  = p; p += (size_t)1024 * 2048;
    bf16* wuqT  = p; p += (size_t)2048 * 1024;
    bf16* wdkvT = p; p += (size_t)512  * 2048;
    bf16* wukT  = p; p += (size_t)1024 * 512;
    bf16* wuvT  = p; p += (size_t)2048 * 512;
    bf16* wkrT  = p; p += (size_t)1024 * 512;
    bf16* woT   = p; p += (size_t)2048 * 2048;
    bf16* cq    = p; p += T * 1024;
    bf16* qb    = p; p += HS;
    bf16* ckv   = p; p += T * 512;
    bf16* kbuf  = p; p += HS;
    bf16* vbuf  = p; p += HS;
    bf16* vtb   = p; p += HS;
    bf16* ao    = p; p += HS;

    dim3 blk(256);

    f2bf_kernel<<<dim3((unsigned)(HS / 256)), blk, 0, stream>>>(hs, hsb, HS);
    wconvT_kernel<<<dim3((2048u * 1024u) / 256u), blk, 0, stream>>>(w_dq,  wdqT,  2048, 1024);
    wconvT_kernel<<<dim3((1024u * 2048u) / 256u), blk, 0, stream>>>(w_uq,  wuqT,  1024, 2048);
    wconvT_kernel<<<dim3((2048u * 512u)  / 256u), blk, 0, stream>>>(w_dkv, wdkvT, 2048, 512);
    wconvT_kernel<<<dim3((512u * 1024u)  / 256u), blk, 0, stream>>>(w_uk,  wukT,  512, 1024);
    wconvT_kernel<<<dim3((512u * 2048u)  / 256u), blk, 0, stream>>>(w_uv,  wuvT,  512, 2048);
    wconvT_kernel<<<dim3((512u * 1024u)  / 256u), blk, 0, stream>>>(w_kr,  wkrT,  512, 1024);
    wconvT_kernel<<<dim3((2048u * 2048u) / 256u), blk, 0, stream>>>(w_o,   woT,   2048, 2048);

    // projections (bf16 WMMA, fp32 accum); workgroup tile 128x128
    gemm_kernel<bf16><<<dim3(32, 1024 / 128), blk, 0, stream>>>(
        hsb, 2048, wdqT, 2048, cq, 1024, 4096, 1024, 2048, 1024, 1024, 0);
    gemm_kernel<bf16><<<dim3(32, 2048 / 128), blk, 0, stream>>>(
        cq, 1024, wuqT, 1024, qb, 2048, 4096, 2048, 1024, 2048, 2048, 0);
    gemm_kernel<bf16><<<dim3(32, 512 / 128), blk, 0, stream>>>(
        hsb, 2048, wdkvT, 2048, ckv, 512, 4096, 512, 2048, 512, 512, 0);
    // k_nope -> k[..., h*128 + 64 + d]
    gemm_kernel<bf16><<<dim3(32, 1024 / 128), blk, 0, stream>>>(
        ckv, 512, wukT, 512, kbuf, 2048, 4096, 1024, 512, 64, 128, 64);
    // k_rope (pre-rotation) -> k[..., h*128 + d]
    gemm_kernel<bf16><<<dim3(32, 1024 / 128), blk, 0, stream>>>(
        ckv, 512, wkrT, 512, kbuf, 2048, 4096, 1024, 512, 64, 128, 0);
    gemm_kernel<bf16><<<dim3(32, 2048 / 128), blk, 0, stream>>>(
        ckv, 512, wuvT, 512, vbuf, 2048, 4096, 2048, 512, 2048, 2048, 0);

    vtrans_kernel<<<dim3((unsigned)(HS / 256)), blk, 0, stream>>>(vbuf, vtb);
    rope_kernel<<<dim3((unsigned)((T * NH_ * 32) / 256)), blk, 0, stream>>>(qb, kbuf);

    mla_attn_kernel<<<dim3(16, 32), blk, 0, stream>>>(qb, kbuf, vtb, ao);

    gemm_kernel<float><<<dim3(32, 2048 / 128), blk, 0, stream>>>(
        ao, 2048, woT, 2048, out, 2048, 4096, 2048, 2048, 2048, 2048, 0);
}
